// BertFlattenForSeqCls_21586505630363
// MI455X (gfx1250) — matrix operand order repaired and checked by reading, not compiled, not run on hardware
//
#include <hip/hip_runtime.h>

typedef __attribute__((ext_vector_type(16))) _Float16 v16h;
typedef __attribute__((ext_vector_type(8)))  _Float16 v8h;
typedef __attribute__((ext_vector_type(8)))  float    v8f;

#define WAVES 8  // waves per block (wave32)

// ---- forced global-address-space accessors (avoid flat_load/flat_store) ----
__device__ __forceinline__ v8h gload_v8h(const _Float16* p) {
  return *(const __attribute__((address_space(1))) v8h*)(unsigned long long)p;
}
__device__ __forceinline__ float gload_f(const float* p) {
  return *(const __attribute__((address_space(1))) float*)(unsigned long long)p;
}
__device__ __forceinline__ void gstore_h(_Float16* p, float v) {
  *(__attribute__((address_space(1))) _Float16*)(unsigned long long)p =
      (_Float16)v;
}

// ---------------------------------------------------------------------------
// Build A-fragment (16x32 f16) from a per-lane pointer that already includes
// row (m = lane%16) and half (8*(lane/16)) offsets: two 16B loads at +0, +16.
// ---------------------------------------------------------------------------
__device__ __forceinline__ v16h make_a(const _Float16* __restrict__ p) {
  v8h lo = gload_v8h(p);
  v8h hi = gload_v8h(p + 16);
  v16h r;
#pragma unroll
  for (int i = 0; i < 8; ++i) { r[i] = lo[i]; r[i + 8] = hi[i]; }
  return r;
}

// ---------------------------------------------------------------------------
// Batched WMMA GEMM:  C[M,N] = epi( A[M,K] * B[K,N] )  (f16 in, f32 accum)
//   EPI: 0 = +bias[n] ; 1 = (+bias[n])*scale[n]+shift[n] ; 2 = relu(+bias[n]);
//        3 = none
//   MT:  16-row tiles per wave (1 or 2). Wave tile = (16*MT) x 64.
//   Double-buffered LDS B stripe shared by the 8 waves. Steady-state loop has
//   NO conditionals (final K-step peeled) so prefetch regs are always defined
//   and the allocator coalesces accumulators in place; one barrier per step.
// ---------------------------------------------------------------------------
template <int EPI, int MT>
__global__ __launch_bounds__(256) void gemm_wmma(
    const _Float16* __restrict__ A, int lda, long aS1, long aS2,
    const _Float16* __restrict__ B, int ldb, long bS1, long bS2,
    _Float16* __restrict__ C, int ldc, long cS1, long cS2,
    const float* __restrict__ bias, const float* __restrict__ scale,
    const float* __restrict__ shift, int K, int batch2) {
  constexpr int BSTR = 72;  // padded LDS row stride in halves (144 B)
  __shared__ __align__(16) _Float16 bs[2][32 * BSTR];  // 2 x 4.5 KB

  const int lane  = threadIdx.x;          // 0..31
  const int wave  = threadIdx.y;          // 0..7
  const int tileM = (blockIdx.x * WAVES + wave) * (16 * MT);
  const int n0    = blockIdx.y * 64;
  const int z     = blockIdx.z;
  const int z1    = z / batch2;
  const int z2    = z - z1 * batch2;
  A += z1 * aS1 + z2 * aS2;
  B += z1 * bS1 + z2 * bS2 + n0;
  C += z1 * cS1 + z2 * cS2;

  const int tid  = wave * 32 + lane;  // 0..255
  const int brow = tid >> 3;          // 0..31 (K row within B tile)
  const int bseg = (tid & 7) * 8;     // 0,8,...,56 (N offset; 16B chunk)
  const int ldsW = brow * BSTR + bseg;

  // per-thread incremental global pointers
  const _Float16* bG = B + (long)brow * ldb + bseg;  // points at tile k=0
  const long bStep = (long)32 * ldb;
  const int am = lane & 15, ah = (lane >> 4) * 8;
  const _Float16* aP[MT];
#pragma unroll
  for (int t = 0; t < MT; ++t)
    aP[t] = A + (long)(tileM + t * 16 + am) * lda + ah;

  v8f zero = {0.f, 0.f, 0.f, 0.f, 0.f, 0.f, 0.f, 0.f};
  v8f acc[MT][4];
#pragma unroll
  for (int t = 0; t < MT; ++t)
#pragma unroll
    for (int j = 0; j < 4; ++j) acc[t][j] = zero;

  // ---- prologue: stage B tile 0 into LDS buf 0; A frags for k=0 into regs
  *(v8h*)(&bs[0][ldsW]) = gload_v8h(bG);
  bG += bStep;
  v16h aCur[MT];
#pragma unroll
  for (int t = 0; t < MT; ++t) { aCur[t] = make_a(aP[t]); aP[t] += 32; }

  const int kSteps = K >> 5;  // always >= 2 here, trip count below >= 1

  // ---- steady state: no conditionals, everything fully defined
#pragma unroll 2
  for (int i = 0; i < kSteps - 1; ++i) {
    const int cur = i & 1;
    __syncthreads();  // bs[cur] stores visible; prior reads of bs[cur^1] done

    // prefetch next tiles while WMMAs run
    v8h bNxt = gload_v8h(bG);
    bG += bStep;
    v16h aNxt[MT];
#pragma unroll
    for (int t = 0; t < MT; ++t) { aNxt[t] = make_a(aP[t]); aP[t] += 32; }

    const _Float16* rp = &bs[cur][0] + (lane >> 4) * 8 * BSTR + (lane & 15);  // lane L = B col n=L%16, k base 8*(L/16)
#pragma unroll
    for (int j = 0; j < 4; ++j) {
      v16h bf;
#pragma unroll
      for (int s = 0; s < 16; ++s) bf[s] = rp[(s < 8 ? s : s + 8) * BSTR + j * 16];
#pragma unroll
      for (int t = 0; t < MT; ++t)
        acc[t][j] = __builtin_amdgcn_wmma_f32_16x16x32_f16(
            false, aCur[t], false, bf, (short)0, acc[t][j], false, false);
    }

    *(v8h*)(&bs[cur ^ 1][ldsW]) = bNxt;
#pragma unroll
    for (int t = 0; t < MT; ++t) aCur[t] = aNxt[t];
  }

  // ---- peeled final K-step (no prefetch, no store)
  {
    const int cur = (kSteps - 1) & 1;
    __syncthreads();
    const _Float16* rp = &bs[cur][0] + (lane >> 4) * 8 * BSTR + (lane & 15);
#pragma unroll
    for (int j = 0; j < 4; ++j) {
      v16h bf;
#pragma unroll
      for (int s = 0; s < 16; ++s) bf[s] = rp[(s < 8 ? s : s + 8) * BSTR + j * 16];
#pragma unroll
      for (int t = 0; t < MT; ++t)
        acc[t][j] = __builtin_amdgcn_wmma_f32_16x16x32_f16(
            false, aCur[t], false, bf, (short)0, acc[t][j], false, false);
    }
  }

  // ---- epilogue
  const int ncl = lane & 15;
  const int mh  = (lane >> 4) * 8;
#pragma unroll
  for (int t = 0; t < MT; ++t) {
#pragma unroll
    for (int j = 0; j < 4; ++j) {
      const int ncol = n0 + j * 16 + ncl;
      const float bi = (EPI == 3) ? 0.f : gload_f(bias + ncol);
      const float sc = (EPI == 1) ? gload_f(scale + ncol) : 1.f;
      const float sh = (EPI == 1) ? gload_f(shift + ncol) : 0.f;
#pragma unroll
      for (int r = 0; r < 8; ++r) {
        float v = acc[t][j][r] + bi;
        v = v * sc + sh;
        if (EPI == 2) v = v > 0.f ? v : 0.f;
        gstore_h(C + (long)(tileM + t * 16 + mh + r) * ldc + ncol, v);
      }
    }
  }
}

// ---------------------------------------------------------------------------
// f32 -> f16 elementwise
// ---------------------------------------------------------------------------
__global__ void cvt_f16_kernel(const float* __restrict__ in,
                               _Float16* __restrict__ out, long n) {
  long i = (long)blockIdx.x * blockDim.x + threadIdx.x;
  const long stride = (long)gridDim.x * blockDim.x;
  for (; i < n; i += stride) gstore_h(out + i, gload_f(in + i));
}

// ---------------------------------------------------------------------------
// f32 [R,C] -> f16 [C,R] transpose (batched over blockIdx.z). Dims % 32 == 0.
// ---------------------------------------------------------------------------
__global__ __launch_bounds__(256) void transpose_cvt_kernel(
    const float* __restrict__ in, _Float16* __restrict__ out, int R, int Cc) {
  __shared__ float t[32][33];
  const long zoff = (long)blockIdx.z * R * Cc;
  in += zoff;
  out += zoff;
  const int c0 = blockIdx.x * 32, r0 = blockIdx.y * 32;
  for (int i = threadIdx.y; i < 32; i += 8)
    t[i][threadIdx.x] = gload_f(in + (long)(r0 + i) * Cc + (c0 + threadIdx.x));
  __syncthreads();
  for (int i = threadIdx.y; i < 32; i += 8)
    gstore_h(out + (long)(c0 + i) * R + (r0 + threadIdx.x),
             t[threadIdx.x][i]);
}

// ---------------------------------------------------------------------------
// logits[b,l] = h[b, t=0, :] . cls_w[l,:] + cls_b[l]
// ---------------------------------------------------------------------------
__global__ void cls_kernel(const _Float16* __restrict__ h,
                           const float* __restrict__ w,
                           const float* __restrict__ b,
                           float* __restrict__ out, int D_, long rowStride,
                           int NL_) {
  __shared__ float red[64];
  const int bi = blockIdx.x, l = blockIdx.y;
  const _Float16* hp = h + (long)bi * rowStride;
  const float* wp = w + (long)l * D_;
  float s = 0.f;
  for (int d = threadIdx.x; d < D_; d += 64) {
    float hv = (float)*(const __attribute__((address_space(1)))
                            _Float16*)(unsigned long long)(hp + d);
    s += hv * gload_f(wp + d);
  }
  red[threadIdx.x] = s;
  __syncthreads();
  for (int st = 32; st > 0; st >>= 1) {
    if (threadIdx.x < st) red[threadIdx.x] += red[threadIdx.x + st];
    __syncthreads();
  }
  if (threadIdx.x == 0) {
    *(__attribute__((address_space(1))) float*)(unsigned long long)(
        out + bi * NL_ + l) = red[0] + gload_f(b + l);
  }
}

// ---------------------------------------------------------------------------
extern "C" void kernel_launch(void* const* d_in, const int* in_sizes, int n_in,
                              void* d_out, int out_size, void* d_ws,
                              size_t ws_size, hipStream_t stream) {
  (void)in_sizes; (void)n_in; (void)out_size; (void)ws_size;
  const int BS = 128, MSL = 128, D = 768, H = 12, DI = 3072, NL = 2;
  const long M = (long)BS * MSL;  // 16384 rows

  const float* x = (const float*)d_in[0];
  const float* Mh[2] = {(const float*)d_in[1], (const float*)d_in[2]};
  const float* P[2][12];
  for (int b = 0; b < 2; ++b)
    for (int i = 0; i < 12; ++i) P[b][i] = (const float*)d_in[3 + b * 12 + i];
  // P[b]: 0 wv_w, 1 wv_b, 2 d0_w, 3 d0_b, 4 lnw, 5 lnb,
  //       6 ff0_w, 7 ff0_b, 8 ff1_w, 9 ff1_b, 10 lnffw, 11 lnffb
  const float* cls_w = (const float*)d_in[27];
  const float* cls_b = (const float*)d_in[28];
  float* outp = (float*)d_out;

  // ---- workspace carve-out (f16 buffers), 256B aligned ----
  char* ws = (char*)d_ws;
  size_t off = 0;
  auto carve = [&](size_t bytes) {
    void* p = ws + off;
    off += (bytes + 255) & ~(size_t)255;
    return (_Float16*)p;
  };
  _Float16* A0 = carve(M * D * 2);   // 24 MB
  _Float16* A1 = carve(M * D * 2);
  _Float16* A2 = carve(M * D * 2);
  _Float16* AU = carve(M * DI * 2);  // 96 MB
  _Float16* WV  = carve((long)D * D * 2);          // Wv^T  [K=D, N=D]
  _Float16* WD  = carve((long)D * D * 2);          // d0^T
  _Float16* WF0 = carve((long)D * DI * 2);         // ff0^T [768, 3072]
  _Float16* WF1 = carve((long)D * DI * 2);         // ff1^T [3072, 768]
  _Float16* MT  = carve((long)H * MSL * MSL * 2);  // M[h]^T

  const dim3 gblk(32, WAVES);

  // input x (squeeze of trailing 1 is a layout no-op) -> f16
  cvt_f16_kernel<<<4096, 256, 0, stream>>>(x, A0, M * D);

  _Float16* hin = A0;
  for (int b = 0; b < 2; ++b) {
    _Float16* V  = (b == 0) ? A1 : A2;
    _Float16* X  = (b == 0) ? A2 : A0;
    _Float16* T1 = (b == 0) ? A0 : A2;
    _Float16* HO = A1;

    // weight transposes for this block (stream-ordered, buffers reused)
    transpose_cvt_kernel<<<dim3(D / 32, D / 32, 1), gblk, 0, stream>>>(
        P[b][0], WV, D, D);
    transpose_cvt_kernel<<<dim3(D / 32, D / 32, 1), gblk, 0, stream>>>(
        P[b][2], WD, D, D);
    transpose_cvt_kernel<<<dim3(D / 32, DI / 32, 1), gblk, 0, stream>>>(
        P[b][6], WF0, DI, D);
    transpose_cvt_kernel<<<dim3(DI / 32, D / 32, 1), gblk, 0, stream>>>(
        P[b][8], WF1, D, DI);
    transpose_cvt_kernel<<<dim3(MSL / 32, MSL / 32, H), gblk, 0, stream>>>(
        Mh[b], MT, MSL, MSL);

    // V = h @ Wv^T + bv          (wave tile 32x64 -> grid.x = M/256)
    gemm_wmma<0, 2><<<dim3(M / 256, D / 64, 1), gblk, 0, stream>>>(
        hin, D, 0, 0, WV, D, 0, 0, V, D, 0, 0,
        P[b][1], nullptr, nullptr, D, 1);

    // sequence mix: X[b,t,h,:] = M[h]^T @ V[b,:,h,:]   (1536 small GEMMs)
    gemm_wmma<3, 1><<<dim3(1, 1, BS * H), gblk, 0, stream>>>(
        MT, MSL, 0, (long)MSL * MSL,
        V, D, (long)MSL * D, 64,
        X, D, (long)MSL * D, 64,
        nullptr, nullptr, nullptr, MSL, H);

    // dense0 + folded-LN affine
    gemm_wmma<1, 2><<<dim3(M / 256, D / 64, 1), gblk, 0, stream>>>(
        X, D, 0, 0, WD, D, 0, 0, T1, D, 0, 0,
        P[b][3], P[b][4], P[b][5], D, 1);

    // FFN up + ReLU
    gemm_wmma<2, 2><<<dim3(M / 256, DI / 64, 1), gblk, 0, stream>>>(
        T1, D, 0, 0, WF0, DI, 0, 0, AU, DI, 0, 0,
        P[b][7], nullptr, nullptr, D, 1);

    // FFN down + folded-LN affine
    gemm_wmma<1, 2><<<dim3(M / 256, D / 64, 1), gblk, 0, stream>>>(
        AU, DI, 0, 0, WF1, D, 0, 0, HO, D, 0, 0,
        P[b][9], P[b][10], P[b][11], DI, 1);

    hin = HO;  // block 2 input / final hidden state lives in A1
  }

  // classifier on [CLS] token (t = 0)
  cls_kernel<<<dim3(BS, NL), 64, 0, stream>>>(
      hin, cls_w, cls_b, outp, D, (long)MSL * D, NL);
}
